// Speech_MultiHeadedAttention_44083544326193
// MI455X (gfx1250) — matrix operand order repaired and checked by reading, not compiled, not run on hardware
//
#include <hip/hip_runtime.h>
#include <hip/hip_bf16.h>

typedef __attribute__((ext_vector_type(16))) __bf16 v16bf;
typedef __attribute__((ext_vector_type(8)))  float  v8f;

#define T_LEN 4096
#define E_DIM 256
#define HEADS 4
#define HD_DIM 64

__device__ __forceinline__ unsigned short bf_bits(float f) {
  unsigned u = __builtin_bit_cast(unsigned, f);
  u += 0x7FFFu + ((u >> 16) & 1u);           // round-to-nearest-even
  return (unsigned short)(u >> 16);
}

struct u4x2 { uint4 lo, hi; };

// A-fragment: 16 contiguous elements p[0..15]  (16B-aligned groups)
__device__ __forceinline__ v16bf load_fragA(const unsigned short* p) {
  u4x2 t;
  t.lo = *(const uint4*)(p);
  t.hi = *(const uint4*)(p + 8);
  return __builtin_bit_cast(v16bf, t);
}
// B-fragment: 16 contiguous elements (32B-aligned)
__device__ __forceinline__ v16bf load_fragB(const unsigned short* p) {
  u4x2 t;
  t.lo = *(const uint4*)(p);
  t.hi = *(const uint4*)(p + 8);
  return __builtin_bit_cast(v16bf, t);
}

// ---------------------------------------------------------------------------
// Kernel 0: bulk fp32 -> bf16 conversion (one pass; removes conversions from
// all GEMM inner loops).  n must be a multiple of 4.
// ---------------------------------------------------------------------------
__global__ void cvt_f32_bf16(const float* __restrict__ src,
                             unsigned short* __restrict__ dst, int n) {
  const int i = (blockIdx.x * blockDim.x + threadIdx.x) * 4;
  if (i < n) {
    const float4 f = *(const float4*)(src + i);
    uint2 o;
    o.x = (unsigned)bf_bits(f.x) | ((unsigned)bf_bits(f.y) << 16);
    o.y = (unsigned)bf_bits(f.z) | ((unsigned)bf_bits(f.w) << 16);
    *(uint2*)(dst + i) = o;
  }
}

// ---------------------------------------------------------------------------
// Kernel 1: QKV projection.  One wave computes a 16x64 strip of the
// (8192 x 768) result: 4 independent accumulators share each A fragment.
// Epilogue scatters into head-split bf16 workspaces (v stored transposed).
// ---------------------------------------------------------------------------
__global__ void qkv_gemm_bf16(const unsigned short* __restrict__ xb,
                              const unsigned short* __restrict__ wb,
                              const float* __restrict__ bias,
                              unsigned short* __restrict__ qws,
                              unsigned short* __restrict__ kws,
                              unsigned short* __restrict__ vwsT) {
  const int tm   = blockIdx.x;        // 512 row tiles (16 tokens)
  const int ng   = blockIdx.y;        // 12 col groups (64 cols)
  const int lane = threadIdx.x;
  const int half = lane >> 4;
  const int l15  = lane & 15;
  const int row0 = tm * 16;
  const int col0 = ng * 64;

  v8f acc[4] = {};
  const unsigned short* arow = xb + (size_t)(row0 + l15) * E_DIM;
#pragma unroll
  for (int c = 0; c < 8; ++c) {
    const v16bf A = load_fragA(arow + c * 32 + 16 * half);
#pragma unroll
    for (int j = 0; j < 4; ++j) {
      const v16bf B = load_fragB(wb + (size_t)(col0 + j * 16 + l15) * E_DIM
                                    + c * 32 + 16 * half);
      acc[j] = __builtin_amdgcn_wmma_f32_16x16x32_bf16(false, A, false, B,
                                                       (short)0, acc[j], false, false);
    }
  }

#pragma unroll
  for (int j = 0; j < 4; ++j) {
    const int   n     = col0 + j * 16 + l15;
    const float bn    = bias[n];
    const float scale = (n < 256) ? 0.125f : 1.0f;   // HD^-0.5 on q (post-bias)
#pragma unroll
    for (int r = 0; r < 8; ++r) {
      const int m     = r + 8 * half;                // C layout
      const int token = row0 + m;
      const int b     = token >> 12;
      const int t     = token & (T_LEN - 1);
      const unsigned short h = bf_bits((acc[j][r] + bn) * scale);
      if (n < 256) {
        const int hh = n >> 6, d = n & 63;
        qws[((size_t)(b * HEADS + hh) * T_LEN + t) * HD_DIM + d] = h;
      } else if (n < 512) {
        const int nn = n - 256, hh = nn >> 6, d = nn & 63;
        kws[((size_t)(b * HEADS + hh) * T_LEN + t) * HD_DIM + d] = h;
      } else {
        const int nn = n - 512, hh = nn >> 6, d = nn & 63;
        vwsT[((size_t)(b * HEADS + hh) * HD_DIM + d) * T_LEN + t] = h;
      }
    }
  }
}

// ---------------------------------------------------------------------------
// Kernel 2: sliding-window attention (W=65), one wave per (head, 16-query tile).
// 80 keys -> 5 score tiles (10 WMMA), half-wave shuffle softmax, P through LDS
// (16x96 zero-padded) -> PV (12 WMMA).
// ---------------------------------------------------------------------------
__global__ void local_attn_bf16(const unsigned short* __restrict__ qws,
                                const unsigned short* __restrict__ kws,
                                const unsigned short* __restrict__ vwsT,
                                unsigned short* __restrict__ ows) {
  __shared__ unsigned short Plds[16 * 96];
  const int tt   = blockIdx.x;        // 256 query tiles
  const int bh   = blockIdx.y;        // 8 head-batches
  const int lane = threadIdx.x;
  const int half = lane >> 4;
  const int l15  = lane & 15;
  const int t0   = tt * 16;
  const int key0 = t0 - 32;

  const unsigned short* qbase = qws  + (size_t)bh * T_LEN * HD_DIM;
  const unsigned short* kbase = kws  + (size_t)bh * T_LEN * HD_DIM;
  const unsigned short* vbase = vwsT + (size_t)bh * HD_DIM * T_LEN;

  // Q fragments for the two 32-wide K-chunks over head dim
  v16bf Aq[2];
  {
    const unsigned short* qrow = qbase + (size_t)(t0 + l15) * HD_DIM;
#pragma unroll
    for (int c = 0; c < 2; ++c) Aq[c] = load_fragA(qrow + c * 32 + 16 * half);
  }

  // scores: S[m][n] = q[t0+m] . k[key0+n], n in 0..79
  v8f sacc[5];
#pragma unroll
  for (int kt = 0; kt < 5; ++kt) {
    v8f acc = {};
    const int  key   = key0 + kt * 16 + l15;       // this lane's key (B column)
    const bool keyok = (key >= 0) && (key < T_LEN);
    const unsigned short* krow = kbase + (size_t)(keyok ? key : 0) * HD_DIM;
#pragma unroll
    for (int c = 0; c < 2; ++c) {
      v16bf Bk = {};
      if (keyok) Bk = load_fragB(krow + c * 32 + 16 * half);  // OOB -> 0 (matches zero-pad)
      acc = __builtin_amdgcn_wmma_f32_16x16x32_bf16(false, Aq[c], false, Bk,
                                                    (short)0, acc, false, false);
    }
    sacc[kt] = acc;
  }

  // window mask + softmax per row (a row's 16 tile-cols live across a lane-half)
#pragma unroll
  for (int r = 0; r < 8; ++r) {
    const int m  = r + 8 * half;
    float     mx = -3.0e38f;
#pragma unroll
    for (int kt = 0; kt < 5; ++kt) {
      const int  nn    = kt * 16 + l15;
      const bool valid = (nn >= m) && (nn <= m + 64);   // |t_k - t_q| <= 32
      const float s    = valid ? sacc[kt][r] : -3.0e38f;
      sacc[kt][r] = s;
      mx = fmaxf(mx, s);
    }
#pragma unroll
    for (int d = 1; d < 16; d <<= 1) mx = fmaxf(mx, __shfl_xor(mx, d, 32));
    float sum = 0.0f;
#pragma unroll
    for (int kt = 0; kt < 5; ++kt) {
      const float s = sacc[kt][r];
      const float p = (s > -1.0e38f) ? __expf(s - mx) : 0.0f;
      sacc[kt][r] = p;
      sum += p;
    }
#pragma unroll
    for (int d = 1; d < 16; d <<= 1) sum += __shfl_xor(sum, d, 32);
    const float inv = 1.0f / sum;
#pragma unroll
    for (int kt = 0; kt < 5; ++kt) sacc[kt][r] *= inv;
  }

  // P -> LDS (16 rows x 96 cols, cols 80..95 zero)
#pragma unroll
  for (int kt = 0; kt < 5; ++kt)
#pragma unroll
    for (int r = 0; r < 8; ++r)
      Plds[(r + 8 * half) * 96 + kt * 16 + l15] = bf_bits(sacc[kt][r]);
  for (int i = lane; i < 256; i += 32)
    Plds[(i >> 4) * 96 + 80 + (i & 15)] = 0;
  __syncthreads();

  // O[m][d] = sum_n P[m][n] * V[key0+n][d], K padded to 96 (3 chunks of 32)
  v8f oacc[4] = {};
#pragma unroll
  for (int c = 0; c < 3; ++c) {
    const v16bf Ap = load_fragA(&Plds[l15 * 96 + c * 32 + 16 * half]);
    const int ks = key0 + c * 32 + 16 * half;      // 16 sequential keys per lane-half
#pragma unroll
    for (int dt = 0; dt < 4; ++dt) {
      const unsigned short* vrow = vbase + (size_t)(dt * 16 + l15) * T_LEN;
      v16bf Bv;
      if (ks >= 0 && ks + 15 < T_LEN) {
        Bv = load_fragB(vrow + ks);
      } else {
#pragma unroll
        for (int i = 0; i < 16; ++i) {
          const int k = ks + i;
          unsigned short hv = (k >= 0 && k < T_LEN) ? vrow[k] : (unsigned short)0;
          Bv[i] = __builtin_bit_cast(__bf16, hv);
        }
      }
      oacc[dt] = __builtin_amdgcn_wmma_f32_16x16x32_bf16(false, Ap, false, Bv,
                                                         (short)0, oacc[dt], false, false);
    }
  }

  // write back in (B, T, H*HD) token-major layout for the output GEMM
  const int b  = bh >> 2;
  const int hh = bh & 3;
#pragma unroll
  for (int dt = 0; dt < 4; ++dt)
#pragma unroll
    for (int r = 0; r < 8; ++r) {
      const int m = r + 8 * half;
      const int d = dt * 16 + l15;
      ows[((size_t)(b * T_LEN + t0 + m)) * E_DIM + hh * HD_DIM + d] = bf_bits(oacc[dt][r]);
    }
}

// ---------------------------------------------------------------------------
// Kernel 3: output projection.  One wave computes a 16x64 strip of the
// (8192 x 256) result; fp32 output + bias.
// ---------------------------------------------------------------------------
__global__ void out_gemm_bf16(const unsigned short* __restrict__ aws,
                              const unsigned short* __restrict__ wb,
                              const float* __restrict__ bias,
                              float* __restrict__ out) {
  const int tm   = blockIdx.x;        // 512
  const int ng   = blockIdx.y;        // 4 col groups of 64
  const int lane = threadIdx.x;
  const int half = lane >> 4;
  const int l15  = lane & 15;
  const int row0 = tm * 16;
  const int col0 = ng * 64;

  v8f acc[4] = {};
  const unsigned short* arow = aws + (size_t)(row0 + l15) * E_DIM;
#pragma unroll
  for (int c = 0; c < 8; ++c) {
    const v16bf A = load_fragA(arow + c * 32 + 16 * half);
#pragma unroll
    for (int j = 0; j < 4; ++j) {
      const v16bf B = load_fragB(wb + (size_t)(col0 + j * 16 + l15) * E_DIM
                                    + c * 32 + 16 * half);
      acc[j] = __builtin_amdgcn_wmma_f32_16x16x32_bf16(false, A, false, B,
                                                       (short)0, acc[j], false, false);
    }
  }
#pragma unroll
  for (int j = 0; j < 4; ++j) {
    const int   n  = col0 + j * 16 + l15;
    const float bn = bias[n];
#pragma unroll
    for (int r = 0; r < 8; ++r)
      out[(size_t)(row0 + r + 8 * half) * E_DIM + n] = acc[j][r] + bn;
  }
}

// ---------------------------------------------------------------------------
extern "C" void kernel_launch(void* const* d_in, const int* in_sizes, int n_in,
                              void* d_out, int out_size, void* d_ws, size_t ws_size,
                              hipStream_t stream) {
  const float* x    = (const float*)d_in[0];
  const float* Wqkv = (const float*)d_in[1];
  const float* bqkv = (const float*)d_in[2];
  const float* Wout = (const float*)d_in[3];
  const float* bout = (const float*)d_in[4];
  float*       out  = (float*)d_out;

  const size_t nX    = (size_t)2 * T_LEN * E_DIM;        // 2,097,152
  const size_t nWqkv = (size_t)3 * E_DIM * E_DIM;        //   196,608
  const size_t nWout = (size_t)E_DIM * E_DIM;            //    65,536
  const size_t nHead = (size_t)2 * HEADS * T_LEN * HD_DIM;

  unsigned short* xb   = (unsigned short*)d_ws;
  unsigned short* wqb  = xb  + nX;
  unsigned short* wob  = wqb + nWqkv;
  unsigned short* qws  = wob + nWout;
  unsigned short* kws  = qws + nHead;
  unsigned short* vws  = kws + nHead;
  unsigned short* ows  = vws + nHead;

  cvt_f32_bf16<<<dim3((nX    / 4 + 255) / 256), 256, 0, stream>>>(x,    xb,  (int)nX);
  cvt_f32_bf16<<<dim3((nWqkv / 4 + 255) / 256), 256, 0, stream>>>(Wqkv, wqb, (int)nWqkv);
  cvt_f32_bf16<<<dim3((nWout / 4 + 255) / 256), 256, 0, stream>>>(Wout, wob, (int)nWout);

  qkv_gemm_bf16<<<dim3(512, 12), 32, 0, stream>>>(xb, wqb, bqkv, qws, kws, vws);
  local_attn_bf16<<<dim3(256, 8), 32, 0, stream>>>(qws, kws, vws, ows);
  out_gemm_bf16<<<dim3(512, 4), 32, 0, stream>>>(ows, wob, bout, out);
}
